// CausalSelfAttention_67405216743831
// MI455X (gfx1250) — matrix operand order repaired and checked by reading, not compiled, not run on hardware
//
#include <hip/hip_runtime.h>
#include <hip/hip_bf16.h>
#include <stdint.h>

#define DIM 1536
#define NH 12
#define HD 128

typedef __attribute__((ext_vector_type(16))) __bf16 v16bf;
typedef __attribute__((ext_vector_type(8)))  float  v8f;

union AFrag {
  v16bf v;
  uint4 q[2];
};

__device__ __forceinline__ unsigned short f2bf(float f) {
  unsigned int u = __float_as_uint(f);
  u += 0x7FFFu + ((u >> 16) & 1u);   // round-to-nearest-even
  return (unsigned short)(u >> 16);
}

// CDNA5 async copy: global -> LDS, no VGPR landing, tracked by ASYNCcnt.
__device__ __forceinline__ void async_copy16(unsigned short* lds_dst,
                                             const unsigned short* gsrc) {
  unsigned off = (unsigned)(uintptr_t)lds_dst;   // low 32 bits = LDS offset
  asm volatile("global_load_async_to_lds_b128 %0, %1, off"
               :: "v"(off), "v"(gsrc)
               : "memory");
}
template <int N>
__device__ __forceinline__ void wait_asyncN() {
  asm volatile("s_wait_asynccnt %0" :: "i"(N) : "memory");
}

// ---------------------------------------------------------------- converts
__global__ void k_cvt_bf16(const float* __restrict__ src,
                           unsigned short* __restrict__ dst, int n) {
  int i = blockIdx.x * blockDim.x + threadIdx.x;
  if (i < n) dst[i] = f2bf(src[i]);
}

// ---------------------------------------------------------------- GEMM
// One block = 32 output rows x all 1536 cols. 12 waves; each wave owns
// 8 N-tiles x 2 M-tiles. Inner loop is software-pipelined: B fragment
// double-buffered per t, A fragments double-buffered per k-step, so each
// WMMA's operands were requested one stage earlier (partial loadcnt waits).
// mode 1: +bias, RMS over full row, RoPE -> outQK bf16 [head][S][128]
// mode 0: +bias -> outVT bf16 [head][128][S]
// mode 2: +bias -> outF fp32 [S][1536]
__global__ __launch_bounds__(384) void k_gemm32(
    const unsigned short* __restrict__ A,
    const unsigned short* __restrict__ Wb,
    const float* __restrict__ bias,
    const float* __restrict__ g,
    float* __restrict__ outF,
    unsigned short* __restrict__ outQK,
    unsigned short* __restrict__ outVT,
    const float* __restrict__ cf, const float* __restrict__ sf,
    const float* __restrict__ ch, const float* __restrict__ sh,
    const float* __restrict__ cw, const float* __restrict__ sw,
    const int* __restrict__ pH, const int* __restrict__ pW,
    int S, int mode) {
  __shared__ float s_red[32][12];
  __shared__ float s_tot[32];
  const int lane = threadIdx.x & 31;
  const int wave = threadIdx.x >> 5;     // 0..11
  const int hl   = lane >> 4;
  const int ln   = lane & 15;
  const int rowBase = blockIdx.x * 32;
  const int nBase = wave * 128;          // 8 tiles of 16 cols

  v8f zero{};
  v8f acc[8][2];
#pragma unroll
  for (int t = 0; t < 8; ++t) { acc[t][0] = zero; acc[t][1] = zero; }

  const unsigned short* arow0 = A + (size_t)(rowBase + ln) * DIM;
  const unsigned short* arow1 = A + (size_t)(rowBase + 16 + ln) * DIM;

  auto loadA = [&](int kk, AFrag& x0, AFrag& x1) {
    x0.q[0] = *(const uint4*)(arow0 + kk + hl * 8);
    x0.q[1] = *(const uint4*)(arow0 + kk + 16 + hl * 8);
    x1.q[0] = *(const uint4*)(arow1 + kk + hl * 8);
    x1.q[1] = *(const uint4*)(arow1 + kk + 16 + hl * 8);
  };
  auto loadB = [&](int kk, int t, AFrag& b) {
    const unsigned short* brow =
        Wb + (size_t)(nBase + t * 16 + ln) * DIM + kk + hl * 8;
    b.q[0] = ((const uint4*)brow)[0];
    b.q[1] = ((const uint4*)brow)[2];
  };

  AFrag a0c, a1c, a0n, a1n, bb[2];
  loadA(0, a0c, a1c);
  loadB(0, 0, bb[0]);
  for (int k0 = 0; k0 < DIM; k0 += 32) {
    const int kN = (k0 + 32 < DIM) ? (k0 + 32) : 0;  // wrap keeps addr valid
#pragma unroll
    for (int t = 0; t < 8; ++t) {
      if (t == 0) loadA(kN, a0n, a1n);
      if (t < 7)  loadB(k0, t + 1, bb[(t + 1) & 1]);
      else        loadB(kN, 0, bb[0]);
      acc[t][0] = __builtin_amdgcn_wmma_f32_16x16x32_bf16(
          false, a0c.v, false, bb[t & 1].v, (short)0, acc[t][0], false, false);
      acc[t][1] = __builtin_amdgcn_wmma_f32_16x16x32_bf16(
          false, a1c.v, false, bb[t & 1].v, (short)0, acc[t][1], false, false);
    }
    a0c = a0n; a1c = a1n;
  }

  // bias (column-indexed)
#pragma unroll
  for (int t = 0; t < 8; ++t) {
    const float bn = bias[nBase + t * 16 + ln];
#pragma unroll
    for (int mt = 0; mt < 2; ++mt)
#pragma unroll
      for (int r = 0; r < 8; ++r) acc[t][mt][r] += bn;
  }

  float rmsf[2][8];
  if (mode == 1) {
    // row sum-of-squares: lane partial -> half-wave shuffle -> cross-wave LDS
#pragma unroll
    for (int mt = 0; mt < 2; ++mt)
#pragma unroll
      for (int r = 0; r < 8; ++r) {
        float ss = 0.f;
#pragma unroll
        for (int t = 0; t < 8; ++t) ss += acc[t][mt][r] * acc[t][mt][r];
#pragma unroll
        for (int off = 1; off < 16; off <<= 1) ss += __shfl_xor(ss, off, 32);
        if (ln == 0) s_red[mt * 16 + r + 8 * hl][wave] = ss;
      }
    __syncthreads();
    if (threadIdx.x < 32) {
      float tot = 0.f;
#pragma unroll
      for (int j = 0; j < 12; ++j) tot += s_red[threadIdx.x][j];
      s_tot[threadIdx.x] = rsqrtf(tot * (1.0f / DIM) + 1e-6f);
    }
    __syncthreads();
#pragma unroll
    for (int mt = 0; mt < 2; ++mt)
#pragma unroll
      for (int r = 0; r < 8; ++r) rmsf[mt][r] = s_tot[mt * 16 + r + 8 * hl];
  }

  const int Hh = (mode == 1) ? *pH : 1;
  const int Ww = (mode == 1) ? *pW : 1;

#pragma unroll
  for (int t = 0; t < 8; ++t) {
    const int j = nBase + t * 16 + ln;   // output column 0..1535
    const int head = j >> 7;
    const int d = j & 127;
#pragma unroll
    for (int mt = 0; mt < 2; ++mt)
#pragma unroll
      for (int r = 0; r < 8; ++r) {
        const int m = mt * 16 + r + 8 * hl;
        const int srow = rowBase + m;
        float val = acc[t][mt][r];
        if (mode == 1) {
          val *= rmsf[mt][r] * g[j];
          // 3-axis RoPE: c<22 -> f table, c<43 -> h table, else w table
          const int c = d >> 1;
          const int par = j & 1;
          const int wIdx = srow % Ww;
          const int tmp = srow / Ww;
          const int hIdx = tmp % Hh;
          const int fIdx = tmp / Hh;
          float co, si;
          if (c < 22)      { co = cf[fIdx * 22 + c];        si = sf[fIdx * 22 + c]; }
          else if (c < 43) { co = ch[hIdx * 21 + (c - 22)]; si = sh[hIdx * 21 + (c - 22)]; }
          else             { co = cw[wIdx * 21 + (c - 43)]; si = sw[wIdx * 21 + (c - 43)]; }
          // pair partner (d^1) lives in adjacent lane
          const float other = __shfl_xor(val, 1, 32);
          val = (par == 0) ? (val * co - other * si) : (other * si + val * co);
          outQK[((size_t)head * S + srow) * HD + d] = f2bf(val);
        } else if (mode == 0) {
          outVT[((size_t)head * HD + d) * S + srow] = f2bf(val);
        } else {
          outF[(size_t)srow * DIM + j] = val;
        }
      }
  }
}

// ---------------------------------------------------------------- flash attn
// 96 threads = 3 waves; block = one head x 96 q-rows (4032/96 = 42 exact).
// Each wave owns 32 q-rows (2 M-tiles). 32-key steps; K/V tiles live in
// double-buffered LDS filled by CDNA5 async global->LDS copies. The next
// tile's copies are issued before computing on the current tile and waited
// with a partial s_wait_asynccnt (12 = exactly one tile's ops per wave).
__global__ __launch_bounds__(96) void k_flash(
    const unsigned short* __restrict__ Qb,   // [NH][S][128] bf16
    const unsigned short* __restrict__ Kb,   // [NH][S][128] bf16
    const unsigned short* __restrict__ Vt,   // [NH][128][S] bf16
    unsigned short* __restrict__ Ab,         // [S][1536] bf16
    int S) {
  __shared__ unsigned short Kt[2][32 * 128];   // [key][d]   2 x 8 KB
  __shared__ unsigned short Vtl[2][128 * 32];  // [d][key]   2 x 8 KB
  __shared__ unsigned short Pl[3 * 32 * 32];   // per-wave P     6 KB
  const int tid = threadIdx.x;
  const int lane = tid & 31;
  const int wave = tid >> 5;                 // 0..2
  const int ln = lane & 15;
  const int hl = lane >> 4;
  const int nQB = S / 96;
  const int qb = blockIdx.x % nQB;
  const int head = blockIdx.x / nQB;
  const int qRow = qb * 96 + wave * 32;
  const float scale = 0.088388347648318447f; // 1/sqrt(128)

  const unsigned short* Qh = Qb + (size_t)head * S * HD;
  const unsigned short* Kh = Kb + (size_t)head * S * HD;
  const unsigned short* Vh = Vt + (size_t)head * HD * S;
  unsigned short* Pw = &Pl[wave * 32 * 32];

  // Stage one 32-key K tile + V tile. Loops are padded to exactly 6
  // iterations per thread (clamped duplicate chunk) so every wave issues
  // exactly 12 async ops -> uniform s_wait_asynccnt immediates.
  auto stage = [&](int buf, int kb) {
#pragma unroll
    for (int it = 0; it < 6; ++it) {
      int i = tid + it * 96;
      const int ii = (i < 512) ? i : 511;
      const int row = ii >> 4, o8 = (ii & 15) * 8;
      async_copy16(&Kt[buf][row * 128 + o8],
                   Kh + (size_t)(kb + row) * HD + o8);
    }
#pragma unroll
    for (int it = 0; it < 6; ++it) {
      int i = tid + it * 96;
      const int ii = (i < 512) ? i : 511;
      const int row = ii >> 2, o8 = (ii & 3) * 8;
      async_copy16(&Vtl[buf][row * 32 + o8],
                   Vh + (size_t)row * S + kb + o8);
    }
  };

  // Q fragments for 2 M-tiles x 4 K-chunks, kept in registers
  AFrag qf[2][4];
#pragma unroll
  for (int mt = 0; mt < 2; ++mt) {
    const unsigned short* qrow = Qh + (size_t)(qRow + mt * 16 + ln) * HD;
#pragma unroll
    for (int c = 0; c < 4; ++c) {
      qf[mt][c].q[0] = *(const uint4*)(qrow + c * 32 + hl * 8);
      qf[mt][c].q[1] = *(const uint4*)(qrow + c * 32 + 16 + hl * 8);
    }
  }

  v8f zero{};
  v8f O[2][8];
  float mrun[2][8], lrun[2][8];
#pragma unroll
  for (int mt = 0; mt < 2; ++mt)
#pragma unroll
    for (int r = 0; r < 8; ++r) {
      O[mt][r] = zero;
      mrun[mt][r] = -1e30f;
      lrun[mt][r] = 0.f;
    }

  stage(0, 0);
  const int NSTEP = S / 32;
  for (int s = 0; s < NSTEP; ++s) {
    const int kb = s * 32;
    const int cur = s & 1;
    if (s + 1 < NSTEP) {          // overlap next tile's fill with this compute
      stage(cur ^ 1, kb + 32);
      wait_asyncN<12>();          // only require the *current* tile landed
    } else {
      wait_asyncN<0>();
    }
    __syncthreads();
    const unsigned short* KtC = &Kt[cur][0];
    const unsigned short* VtC = &Vtl[cur][0];

    // ---- scores: 16 WMMAs (2 mt x 2 key-halves x 4 K-chunks)
    v8f sc[2][2];
    sc[0][0] = zero; sc[0][1] = zero; sc[1][0] = zero; sc[1][1] = zero;
#pragma unroll
    for (int c = 0; c < 4; ++c) {
      AFrag b0, b1;
      b0.q[0] = *(const uint4*)(&KtC[ln * 128 + c * 32 + hl * 8]);
      b0.q[1] = *(const uint4*)(&KtC[ln * 128 + c * 32 + 16 + hl * 8]);
      b1.q[0] = *(const uint4*)(&KtC[(16 + ln) * 128 + c * 32 + hl * 8]);
      b1.q[1] = *(const uint4*)(&KtC[(16 + ln) * 128 + c * 32 + 16 + hl * 8]);
#pragma unroll
      for (int mt = 0; mt < 2; ++mt) {
        sc[mt][0] = __builtin_amdgcn_wmma_f32_16x16x32_bf16(
            false, qf[mt][c].v, false, b0.v, (short)0, sc[mt][0], false, false);
        sc[mt][1] = __builtin_amdgcn_wmma_f32_16x16x32_bf16(
            false, qf[mt][c].v, false, b1.v, (short)0, sc[mt][1], false, false);
      }
    }

    // ---- online softmax (row stats across each 16-lane half)
#pragma unroll
    for (int mt = 0; mt < 2; ++mt)
#pragma unroll
      for (int r = 0; r < 8; ++r) {
        float v0 = sc[mt][0][r] * scale;
        float v1 = sc[mt][1][r] * scale;
        float mx = fmaxf(v0, v1);
#pragma unroll
        for (int off = 1; off < 16; off <<= 1)
          mx = fmaxf(mx, __shfl_xor(mx, off, 32));
        float mnew = fmaxf(mrun[mt][r], mx);
        float alpha = __expf(mrun[mt][r] - mnew);
        float p0 = __expf(v0 - mnew);
        float p1 = __expf(v1 - mnew);
        float rs = p0 + p1;
#pragma unroll
        for (int off = 1; off < 16; off <<= 1) rs += __shfl_xor(rs, off, 32);
        lrun[mt][r] = lrun[mt][r] * alpha + rs;
        mrun[mt][r] = mnew;
#pragma unroll
        for (int cc = 0; cc < 8; ++cc) O[mt][cc][r] *= alpha;
        const int m = mt * 16 + r + 8 * hl;
        Pw[m * 32 + ln] = f2bf(p0);
        Pw[m * 32 + 16 + ln] = f2bf(p1);
      }

    // ---- P*V: per-wave P (LDS round-trip re-fragments it as A-matrix)
#pragma unroll
    for (int mt = 0; mt < 2; ++mt) {
      AFrag pf;
      pf.q[0] = *(const uint4*)(&Pw[(mt * 16 + ln) * 32 + hl * 8]);
      pf.q[1] = *(const uint4*)(&Pw[(mt * 16 + ln) * 32 + 16 + hl * 8]);
#pragma unroll
      for (int cc = 0; cc < 8; ++cc) {
        AFrag vf;
        vf.q[0] = *(const uint4*)(&VtC[(cc * 16 + ln) * 32 + hl * 8]);
        vf.q[1] = *(const uint4*)(&VtC[(cc * 16 + ln) * 32 + 16 + hl * 8]);
        O[mt][cc] = __builtin_amdgcn_wmma_f32_16x16x32_bf16(
            false, pf.v, false, vf.v, (short)0, O[mt][cc], false, false);
      }
    }
    __syncthreads();   // everyone done reading buf before it's restaged
  }

#pragma unroll
  for (int mt = 0; mt < 2; ++mt)
#pragma unroll
    for (int cc = 0; cc < 8; ++cc)
#pragma unroll
      for (int r = 0; r < 8; ++r) {
        const int m = mt * 16 + r + 8 * hl;
        const int srow = qRow + m;
        const int d = cc * 16 + ln;
        Ab[(size_t)srow * DIM + head * HD + d] = f2bf(O[mt][cc][r] / lrun[mt][r]);
      }
}

// ---------------------------------------------------------------- launch
extern "C" void kernel_launch(void* const* d_in, const int* in_sizes, int n_in,
                              void* d_out, int out_size, void* d_ws, size_t ws_size,
                              hipStream_t stream) {
  const float* x  = (const float*)d_in[0];
  const float* Wq = (const float*)d_in[1];
  const float* bq = (const float*)d_in[2];
  const float* Wk = (const float*)d_in[3];
  const float* bk = (const float*)d_in[4];
  const float* Wv = (const float*)d_in[5];
  const float* bv = (const float*)d_in[6];
  const float* Wo = (const float*)d_in[7];
  const float* bo = (const float*)d_in[8];
  const float* gq = (const float*)d_in[9];
  const float* gk = (const float*)d_in[10];
  const float* cf = (const float*)d_in[11];
  const float* sf = (const float*)d_in[12];
  const float* ch = (const float*)d_in[13];
  const float* sh = (const float*)d_in[14];
  const float* cw = (const float*)d_in[15];
  const float* sw = (const float*)d_in[16];
  const int* pH = (const int*)d_in[18];
  const int* pW = (const int*)d_in[19];
  (void)n_in; (void)ws_size; (void)out_size;
  const int S = in_sizes[0] / DIM;   // 4032

  char* w = (char*)d_ws;
  size_t off = 0;
  auto take = [&](size_t bytes) -> char* {
    char* p = w + off;
    off += (bytes + 255) & ~(size_t)255;
    return p;
  };
  unsigned short* Xb  = (unsigned short*)take((size_t)S * DIM * 2);
  unsigned short* Wqb = (unsigned short*)take((size_t)DIM * DIM * 2);
  unsigned short* Wkb = (unsigned short*)take((size_t)DIM * DIM * 2);
  unsigned short* Wvb = (unsigned short*)take((size_t)DIM * DIM * 2);
  unsigned short* Wob = (unsigned short*)take((size_t)DIM * DIM * 2);
  unsigned short* Qr  = (unsigned short*)take((size_t)NH * S * HD * 2);
  unsigned short* Kr  = (unsigned short*)take((size_t)NH * S * HD * 2);
  unsigned short* Vtb = (unsigned short*)take((size_t)NH * HD * S * 2);
  unsigned short* Ab  = (unsigned short*)take((size_t)S * DIM * 2);

  const int nX = S * DIM, nW = DIM * DIM;
  k_cvt_bf16<<<(nX + 255) / 256, 256, 0, stream>>>(x,  Xb,  nX);
  k_cvt_bf16<<<(nW + 255) / 256, 256, 0, stream>>>(Wq, Wqb, nW);
  k_cvt_bf16<<<(nW + 255) / 256, 256, 0, stream>>>(Wk, Wkb, nW);
  k_cvt_bf16<<<(nW + 255) / 256, 256, 0, stream>>>(Wv, Wvb, nW);
  k_cvt_bf16<<<(nW + 255) / 256, 256, 0, stream>>>(Wo, Wob, nW);

  dim3 gg(S / 32);
  // Q: bias + RMS + RoPE
  k_gemm32<<<gg, 384, 0, stream>>>(Xb, Wqb, bq, gq, nullptr, Qr, nullptr,
                                   cf, sf, ch, sh, cw, sw, pH, pW, S, 1);
  // K: bias + RMS + RoPE
  k_gemm32<<<gg, 384, 0, stream>>>(Xb, Wkb, bk, gk, nullptr, Kr, nullptr,
                                   cf, sf, ch, sh, cw, sw, pH, pW, S, 1);
  // V: bias only, transposed per head
  k_gemm32<<<gg, 384, 0, stream>>>(Xb, Wvb, bv, gq, nullptr, nullptr, Vtb,
                                   cf, sf, ch, sh, cw, sw, pH, pW, S, 0);

  k_flash<<<NH * (S / 96), 96, 0, stream>>>(Qr, Kr, Vtb, Ab, S);

  // output projection, fp32 result straight to d_out
  k_gemm32<<<gg, 384, 0, stream>>>(Ab, Wob, bo, gq, (float*)d_out, nullptr, nullptr,
                                   cf, sf, ch, sh, cw, sw, pH, pW, S, 2);
}
